// DIEN_28638841930519
// MI455X (gfx1250) — matrix operand order repaired and checked by reading, not compiled, not run on hardware
//
#include <hip/hip_runtime.h>
#include <hip/hip_bf16.h>
#include <cstdint>

// ---------------------------------------------------------------------------
// AUGRU (DIEN) scan for gfx1250.  B=256, T=512, D=H=512.
//   Gx precompute:  x_t @ W_x + b  for all t (parallel, bf16 WMMA)
//   Recurrent scan: r/u = sigmoid(h@W_h + Gx), hhat = tanh((h*r)@Wh_h + Gxh)
//                   h   = (1-a*u)*h + a*u*hhat
// Weights DMA'd into LDS via the Tensor Data Mover as pre-swizzled WMMA
// B-fragments; h / h*r exchanged via L2 in WMMA A-fragment order;
// 2 grid barriers per timestep.
// ---------------------------------------------------------------------------

typedef __bf16 bf16;
typedef bf16     bf16x16 __attribute__((ext_vector_type(16)));
typedef float    f32x8   __attribute__((ext_vector_type(8)));
typedef unsigned u32x4   __attribute__((ext_vector_type(4)));
typedef int      i32x4   __attribute__((ext_vector_type(4)));
typedef int      i32x8   __attribute__((ext_vector_type(8)));

#define B_   256
#define T_   512
#define D_   512
#define H_   512
#define TC   64                 // timesteps per chunk
#define NCHUNK (T_ / TC)        // 8
#define BR   32                 // batch rows per block
#define HC   64                 // H columns per block
#define GB   (B_ / BR)          // 8 batch groups
#define GH   (H_ / HC)          // 8 H slices
#define NBLK (GB * GH)          // 64 blocks in the recurrent grid
#define KT_H (H_ / 32)          // 16 k-tiles over h / h*r
#define KT_D (D_ / 32)          // 16 k-tiles over x

// workspace layout (bytes); assumes ws_size >= ~101 MB
#define WS_WPACK   0                            // 3 MB  bf16 packed weights
#define WS_HBUF    (3u << 20)                   // 256 KB bf16 h   (A-frag order)
#define WS_HRBUF   ((3u << 20) + (256u << 10))  // 256 KB bf16 h*r
#define WS_CTR     ((3u << 20) + (512u << 10))  // sync counters
#define WS_GX      (4u << 20)                   // 96 MB fp32 Gx chunk

__device__ __forceinline__ f32x8 wmma_bf16(bf16x16 a, bf16x16 b, f32x8 c) {
  return __builtin_amdgcn_wmma_f32_16x16x32_bf16(false, a, false, b,
                                                 (short)0, c, false, false);
}

__device__ __forceinline__ float sigmoidf_(float x) {
  return 1.f / (1.f + __expf(-x));
}

// grid-wide barrier: monotonically increasing counter, all blocks resident.
__device__ __forceinline__ void grid_sync(unsigned* ctr, unsigned target) {
  __threadfence();
  __syncthreads();
  if (threadIdx.x == 0) {
    atomicAdd(ctr, 1u);
    while (*(volatile unsigned*)ctr < target) __builtin_amdgcn_s_sleep(1);
    __threadfence();
  }
  __syncthreads();
}

// ---------------------------------------------------------------------------
// Tensor Data Mover: DMA a 2D tile (rows x tile_w bytes, row stride in bytes)
// from global memory into LDS at lds_addr.  1-byte elements; D# per CDNA5 ISA
// §8.3/8.4 (group0: count/lds/global/type=2; group1: dims+strides).
// ---------------------------------------------------------------------------
__device__ __forceinline__ void tdm_load_2d(unsigned lds_addr, const void* gptr,
                                            unsigned tile_w_bytes, unsigned rows,
                                            unsigned row_stride_bytes) {
  uint64_t ga = (uint64_t)(uintptr_t)gptr;
  u32x4 g0;
  g0[0] = 1u;                                   // count=1 valid descriptor
  g0[1] = lds_addr;                             // LDS byte address
  g0[2] = (unsigned)ga;                         // global addr [31:0]
  g0[3] = (unsigned)(ga >> 32) | (2u << 30);    // addr [56:32] | type=2
  i32x8 g1;
  g1[0] = 0;                                    // wg_mask=0, data_size=1B
  g1[1] = (int)((row_stride_bytes & 0xFFFFu) << 16);        // tensor_dim0 lo16
  g1[2] = (int)((row_stride_bytes >> 16) | (rows << 16));   // dim0 hi | dim1 lo
  g1[3] = (int)(tile_w_bytes << 16);            // dim1 hi(=0) | tile_dim0
  g1[4] = (int)rows;                            // tile_dim1 | tile_dim2=0
  g1[5] = (int)row_stride_bytes;                // tensor_dim0_stride lo32
  g1[6] = 0;                                    // stride0 hi | stride1 lo
  g1[7] = 0;                                    // stride1 hi
  i32x4 z4 = {0, 0, 0, 0};
#if __clang_major__ >= 23
  i32x8 z8 = {0, 0, 0, 0, 0, 0, 0, 0};
  __builtin_amdgcn_tensor_load_to_lds(g0, g1, z4, z4, z8, 0);
#else
  __builtin_amdgcn_tensor_load_to_lds(g0, g1, z4, z4, 0);
#endif
}

// ---------------------------------------------------------------------------
// init: zero h state + sync counters (re-run every launch / graph replay)
// ---------------------------------------------------------------------------
__global__ void init_ws_kernel(unsigned* hbuf_u32, unsigned* ctr) {
  int i = blockIdx.x * 256 + threadIdx.x;
  if (i < (int)((256u << 10) / 4)) hbuf_u32[i] = 0;   // hbuf bf16 zeros
  if (i < NCHUNK) ctr[i] = 0;
}

// ---------------------------------------------------------------------------
// pack weights fp32 -> bf16 in WMMA B-fragment order, p (h/x half) OUTERMOST
// so each H-slice's h-part is one uniformly strided 2D region (TDM-friendly):
//   wpack[(((p*3+g)*16 + kt)*32 + ct)][lane][e],  K = p*512 + kt*32 + (e<8 ? 8*lhi + e : 8 + 8*lhi + e)
// ---------------------------------------------------------------------------
__global__ void pack_w_kernel(const float* __restrict__ Wr,
                              const float* __restrict__ Wu,
                              const float* __restrict__ Wh,
                              bf16* __restrict__ wp) {
  int idx = blockIdx.x * 256 + threadIdx.x;       // 98304 total
  int L   = idx & 31;
  int ct  = (idx >> 5) & 31;
  int kt  = (idx >> 10) & 15;
  int gp  = idx >> 14;                            // p*3 + g, 0..5
  if (gp >= 6) return;
  int p = gp / 3, g = gp % 3;
  const float* W = (g == 0) ? Wr : (g == 1) ? Wu : Wh;
  int n   = ct * 16 + (L & 15);
  int lhi = L >> 4;
  bf16* dst = wp + (size_t)idx * 16;
#pragma unroll
  for (int e = 0; e < 16; ++e) {
    int k = p * 512 + kt * 32 + (e < 8 ? lhi * 8 + e : 16 + lhi * 8 + (e - 8));     // B-fragment K map
    dst[e] = (bf16)W[(size_t)k * H_ + n];
  }
}

// ---------------------------------------------------------------------------
// Gx precompute for one T-chunk: rows are t-major (row = tloc*256 + b) so each
// 16x16 C-tile is 16 batch rows at a single t — exactly what the scan consumes.
// Output stored in C-fragment order (32B contiguous per lane).
// ---------------------------------------------------------------------------
__global__ void gx_gemm_kernel(const float* __restrict__ X,
                               const bf16* __restrict__ wp,
                               const float* __restrict__ br,
                               const float* __restrict__ bu,
                               const float* __restrict__ bh,
                               float* __restrict__ gx, int t0) {
  const int wid  = threadIdx.x >> 5;
  const int lane = threadIdx.x & 31;
  const int tile = blockIdx.x * 8 + wid;          // 3*1024*32 = 98304 tiles
  const int g    = tile / (1024 * 32);
  const int r    = tile % (1024 * 32);
  const int mt   = r >> 5;                        // 0..1023 M-tiles
  const int ct   = r & 31;                        // 0..31  N-tiles
  const int tloc  = mt >> 4;                      // 0..63
  const int btile = mt & 15;                      // 0..15
  const int lhi = lane >> 4;
  const int Ml  = lane & 15;

  const float* bias = (g == 0) ? br : (g == 1) ? bu : bh;
  const float bv = bias[ct * 16 + Ml];
  f32x8 acc;
#pragma unroll
  for (int j = 0; j < 8; ++j) acc[j] = bv;

  // A rows: batch b = btile*16 + Ml, time t0+tloc
  const float* xrow = X + ((size_t)(btile * 16 + Ml) * T_ + (t0 + tloc)) * D_;
  // x-half of gate g lives at block (1*3 + g) in the packed layout
  const bf16* wxb = wp + ((size_t)(3 + g) * 16) * 32 * 32 * 16;

  for (int kt = 0; kt < KT_D; ++kt) {
    const float* px = xrow + kt * 32 + lhi * 8;   // two contiguous 8-float runs
    float4 a0 = *(const float4*)(px);
    float4 a1 = *(const float4*)(px + 4);
    float4 a2 = *(const float4*)(px + 16);
    float4 a3 = *(const float4*)(px + 20);
    bf16x16 A;
    A[0]=(bf16)a0.x;  A[1]=(bf16)a0.y;  A[2]=(bf16)a0.z;  A[3]=(bf16)a0.w;
    A[4]=(bf16)a1.x;  A[5]=(bf16)a1.y;  A[6]=(bf16)a1.z;  A[7]=(bf16)a1.w;
    A[8]=(bf16)a2.x;  A[9]=(bf16)a2.y;  A[10]=(bf16)a2.z; A[11]=(bf16)a2.w;
    A[12]=(bf16)a3.x; A[13]=(bf16)a3.y; A[14]=(bf16)a3.z; A[15]=(bf16)a3.w;
    bf16x16 Bm = *(const bf16x16*)(wxb + ((size_t)(kt * 32 + ct) * 32 + lane) * 16);
    acc = wmma_bf16(A, Bm, acc);
  }
  float* dst = gx + ((size_t)(((tloc * 3 + g) * 16 + btile) * 32 + ct) * 32 + lane) * 8;
  *(f32x8*)dst = acc;
}

// ---------------------------------------------------------------------------
// Recurrent scan over one T-chunk. 64 blocks x 256 threads (8 waves).
// Per wave: one 16x16 output tile; weights TDM'd into LDS; 2 grid syncs/step.
// ---------------------------------------------------------------------------
__global__ void augru_scan_kernel(const float* __restrict__ att,
                                  const bf16* __restrict__ wp,
                                  const float* __restrict__ gx,
                                  bf16* __restrict__ hbuf,
                                  bf16* __restrict__ hrbuf,
                                  unsigned* __restrict__ ctr,
                                  float* __restrict__ out, int t0) {
  extern __shared__ char smem[];
  bf16*  lw    = (bf16*)smem;                       // 3*16*4*512 bf16 = 192 KB
  bf16*  st    = (bf16*)(smem + 196608);            // 2*16*64 bf16 tile stage
  float* att_s = (float*)(smem + 196608 + 4096);    // 32 attention scores

  const int tid  = threadIdx.x;
  const int wid  = tid >> 5, lane = tid & 31;
  const int bg   = blockIdx.x / GH, hs = blockIdx.x % GH;
  const int mt   = wid >> 2, nt = wid & 3;          // 2 row-chunks x 4 col-tiles
  const int lhi  = lane >> 4, Ml = lane & 15;
  const int rc   = bg * 2 + mt;                     // global 16-row chunk 0..15
  const int r0   = bg * BR + mt * 16;
  const int c0   = hs * HC + nt * 16;
  const int ctile = hs * 4 + nt;

  // ---- TDM: one DMA pulls this block's whole 192 KB h-weight slice into LDS:
  // 48 rows (3 gates x 16 k-tiles) of 4 KB (4 ct x 32 lanes x 16 bf16) at a
  // uniform 32 KB row stride.  Waves overlap descriptor + h_c reconstruction.
  if (wid == 0) {
    tdm_load_2d(__builtin_amdgcn_groupstaticsize(),           // LDS base of lw
                wp + (size_t)hs * 4 * 32 * 16,                // + hs*4096 bytes
                4096u, 48u, 32768u);
  }

  // ---- reconstruct fp32 C-layout h tile from hbuf (zeros on chunk 0)
  f32x8 h_c;
#pragma unroll
  for (int j = 0; j < 8; ++j) {
    int M = j + 8 * lhi;
    int col = c0 + Ml;
    int kt = col >> 5, Kf = col & 31;
    int lh2, e2;
    if (Kf < 8)       { lh2 = 0; e2 = Kf; }
    else if (Kf < 16) { lh2 = 1; e2 = Kf - 8; }
    else if (Kf < 24) { lh2 = 0; e2 = Kf - 8; }
    else              { lh2 = 1; e2 = Kf - 16; }
    h_c[j] = (float)hbuf[(((size_t)rc * 16 + kt) * 32 + (M + 16 * lh2)) * 16 + e2];
  }

  if (wid == 0) __builtin_amdgcn_s_wait_tensorcnt(0);   // weights landed in LDS
  __syncthreads();

  unsigned tgt = 0;
  for (int tl = 0; tl < TC; ++tl) {
    const int t = t0 + tl;
    if (tid < 32) att_s[tid] = att[(size_t)(bg * BR + tid) * T_ + t];

    // pre-activation accumulators seeded with Gx (bias + x@W_x), fp32
    const size_t gxb = ((size_t)(tl * 3) * 16 + rc) * 32;
    f32x8 acc_r = *(const f32x8*)(gx + ((gxb + 0 * 16 * 32) + ctile) * 256 + lane * 8);
    f32x8 acc_u = *(const f32x8*)(gx + ((gxb + 1 * 16 * 32) + ctile) * 256 + lane * 8);
    f32x8 acc_h = *(const f32x8*)(gx + ((gxb + 2 * 16 * 32) + ctile) * 256 + lane * 8);
    if (tl + 1 < TC)  // prefetch next step's Gx r-tile (global_prefetch_b8)
      __builtin_prefetch(gx + ((((size_t)(tl + 1) * 3) * 16 + rc) * 32 + ctile) * 256 + lane * 8, 0, 0);

    // ---- phase 1: r,u over the h half of K
    for (int kt = 0; kt < KT_H; ++kt) {
      bf16x16 A  = *(const bf16x16*)(hbuf + (((size_t)rc * 16 + kt) * 32 + lane) * 16);
      bf16x16 Br = *(const bf16x16*)(lw + (((size_t)(0 * 16 + kt) * 4 + nt) * 32 + lane) * 16);
      bf16x16 Bu = *(const bf16x16*)(lw + (((size_t)(1 * 16 + kt) * 4 + nt) * 32 + lane) * 16);
      acc_r = wmma_bf16(A, Br, acc_r);
      acc_u = wmma_bf16(A, Bu, acc_u);
    }
    f32x8 uv, hrv;
#pragma unroll
    for (int j = 0; j < 8; ++j) {
      float rr = sigmoidf_(acc_r[j]);
      uv[j]  = sigmoidf_(acc_u[j]);
      hrv[j] = h_c[j] * rr;
    }

    // ---- publish h*r tile in A-fragment order (LDS bounce -> global)
#pragma unroll
    for (int j = 0; j < 8; ++j)
      st[((size_t)mt * 16 + (j + 8 * lhi)) * 64 + nt * 16 + Ml] = (bf16)hrv[j];
    __syncthreads();
    if (nt < 2) {
      bf16x16 v;
#pragma unroll
      for (int e = 0; e < 16; ++e) {
        int K = e + 8 * ((e >= 8) + lhi);
        v[e] = st[((size_t)mt * 16 + Ml) * 64 + nt * 32 + K];
      }
      *(bf16x16*)(hrbuf + (((size_t)rc * 16 + (hs * 2 + nt)) * 32 + lane) * 16) = v;
    }
    grid_sync(ctr, tgt += NBLK);

    // ---- phase 2: h_hat over the (h*r) half of K
    for (int kt = 0; kt < KT_H; ++kt) {
      bf16x16 A  = *(const bf16x16*)(hrbuf + (((size_t)rc * 16 + kt) * 32 + lane) * 16);
      bf16x16 Bh = *(const bf16x16*)(lw + (((size_t)(2 * 16 + kt) * 4 + nt) * 32 + lane) * 16);
      acc_h = wmma_bf16(A, Bh, acc_h);
    }
    f32x8 hn;
#pragma unroll
    for (int j = 0; j < 8; ++j) {
      float hh = tanhf(acc_h[j]);
      float a  = att_s[mt * 16 + (j + 8 * lhi)];
      float uu = a * uv[j];
      hn[j] = (1.f - uu) * h_c[j] + uu * hh;
    }
    h_c = hn;

    // ---- emit output (fp32) and publish new h in A-fragment order
#pragma unroll
    for (int j = 0; j < 8; ++j) {
      int M = j + 8 * lhi;
      out[((size_t)(r0 + M) * T_ + t) * H_ + c0 + Ml] = hn[j];
      st[((size_t)mt * 16 + M) * 64 + nt * 16 + Ml] = (bf16)hn[j];
    }
    __syncthreads();
    if (nt < 2) {
      bf16x16 v;
#pragma unroll
      for (int e = 0; e < 16; ++e) {
        int K = e + 8 * ((e >= 8) + lhi);
        v[e] = st[((size_t)mt * 16 + Ml) * 64 + nt * 32 + K];
      }
      *(bf16x16*)(hbuf + (((size_t)rc * 16 + (hs * 2 + nt)) * 32 + lane) * 16) = v;
    }
    grid_sync(ctr, tgt += NBLK);
  }
}

// ---------------------------------------------------------------------------
extern "C" void kernel_launch(void* const* d_in, const int* in_sizes, int n_in,
                              void* d_out, int out_size, void* d_ws, size_t ws_size,
                              hipStream_t stream) {
  const float* X   = (const float*)d_in[0];
  const float* att = (const float*)d_in[1];
  const float* Wr  = (const float*)d_in[2];
  const float* br  = (const float*)d_in[3];
  const float* Wu  = (const float*)d_in[4];
  const float* bu  = (const float*)d_in[5];
  const float* Wh  = (const float*)d_in[6];
  const float* bh  = (const float*)d_in[7];
  float* out = (float*)d_out;

  char* ws = (char*)d_ws;                 // assumes ws_size >= ~101 MB
  bf16*     wpack = (bf16*)(ws + WS_WPACK);
  bf16*     hbuf  = (bf16*)(ws + WS_HBUF);
  bf16*     hrbuf = (bf16*)(ws + WS_HRBUF);
  unsigned* ctr   = (unsigned*)(ws + WS_CTR);
  float*    gx    = (float*)(ws + WS_GX);

  init_ws_kernel<<<256, 256, 0, stream>>>((unsigned*)hbuf, ctr);
  pack_w_kernel<<<384, 256, 0, stream>>>(Wr, Wu, Wh, wpack);

  const size_t lds_bytes = 196608 + 4096 + 256;   // weights + stage + att
  for (int c = 0; c < NCHUNK; ++c) {
    gx_gemm_kernel<<<12288, 256, 0, stream>>>(X, wpack, br, bu, bh, gx, c * TC);
    augru_scan_kernel<<<NBLK, 256, lds_bytes, stream>>>(att, wpack, gx, hbuf,
                                                        hrbuf, ctr + c, out, c * TC);
  }
}
